// xLSTMBlock_45887430590547
// MI455X (gfx1250) — matrix operand order repaired and checked by reading, not compiled, not run on hardware
//
#include <hip/hip_runtime.h>
#include <hip/hip_fp16.h>

#define DM 1024
#define DH 512
#define B_ 8
#define T_ 2048
#define BT (B_*T_)
#define LCH 32                          // recurrence chunk length
#define NCH (T_/LCH)
#define EPSV 1e-6f
#define QK_SCALE 0.04419417382415922f   // 1/sqrt(512)

typedef _Float16 half_t;
typedef __attribute__((ext_vector_type(16))) _Float16 v16h;
typedef __attribute__((ext_vector_type(8)))  _Float16 v8h;
typedef __attribute__((ext_vector_type(8)))  float    v8f;
typedef __attribute__((ext_vector_type(4)))  unsigned v4u;
typedef __attribute__((ext_vector_type(8)))  unsigned v8u;

#define WMMA_F16(a, b, c) __builtin_amdgcn_wmma_f32_16x16x32_f16( \
    false, (a), false, (b), (short)0, (c), false, false)

// ---------------------------------------------------------------------------
// CDNA5 async global->LDS copy (ASYNCcnt-tracked, per-lane b128)
// ---------------------------------------------------------------------------
__device__ inline void async_ld_b128(void* lds_dst, const void* gsrc) {
  asm volatile("global_load_async_to_lds_b128 %0, %1, off"
               :: "v"((unsigned)(uintptr_t)lds_dst),
                  "v"((unsigned long long)(uintptr_t)gsrc)
               : "memory");
}
__device__ inline void wait_async_le(int n) {
  if (n == 0) {
#if __has_builtin(__builtin_amdgcn_s_wait_asynccnt)
    __builtin_amdgcn_s_wait_asynccnt(0);
#else
    asm volatile("s_wait_asynccnt 0" ::: "memory");
#endif
  } else {
#if __has_builtin(__builtin_amdgcn_s_wait_asynccnt)
    __builtin_amdgcn_s_wait_asynccnt(8);
#else
    asm volatile("s_wait_asynccnt 8" ::: "memory");
#endif
  }
}
__device__ inline void wait_tensor_le(int n) {
  if (n == 0) {
#if __has_builtin(__builtin_amdgcn_s_wait_tensorcnt)
    __builtin_amdgcn_s_wait_tensorcnt(0);
#else
    asm volatile("s_wait_tensorcnt 0" ::: "memory");
#endif
  } else {
#if __has_builtin(__builtin_amdgcn_s_wait_tensorcnt)
    __builtin_amdgcn_s_wait_tensorcnt(1);
#else
    asm volatile("s_wait_tensorcnt 1" ::: "memory");
#endif
  }
}

// ---------------------------------------------------------------------------
// Tensor Data Mover: 2D f16 tile (rows x row_elems, rows contiguous with
// stride row_elems) global -> LDS. D# bit layout per CDNA5 ISA.
// Issue once per wave (TDM ignores EXEC); tracked by TENSORcnt.
// ---------------------------------------------------------------------------
__device__ inline void tdm_load_2d_f16(unsigned lds_addr, const void* gaddr,
                                       unsigned rows, unsigned row_elems) {
  unsigned long long ga = (unsigned long long)(uintptr_t)gaddr;
  v4u g0;
  g0[0] = 1u;                                        // count=1, user mode
  g0[1] = lds_addr;                                  // LDS byte address
  g0[2] = (unsigned)(ga & 0xffffffffu);              // global_addr[31:0]
  g0[3] = (unsigned)((ga >> 32) & 0x01ffffffu)       // global_addr[56:32]
          | 0x80000000u;                             // type=2 ("image")
  v8u g1;
  g1[0] = 1u << 16;                                  // data_size=1 (2 bytes)
  g1[1] = (row_elems & 0xffffu) << 16;               // tensor_dim0 lo16
  g1[2] = ((row_elems >> 16) & 0xffffu)              // tensor_dim0 hi16
          | ((rows & 0xffffu) << 16);                // tensor_dim1 lo16
  g1[3] = ((rows >> 16) & 0xffffu)                   // tensor_dim1 hi16
          | ((row_elems & 0xffffu) << 16);           // tile_dim0
  g1[4] = rows & 0xffffu;                            // tile_dim1 (tile_dim2=0)
  g1[5] = row_elems;                                 // tensor_dim0_stride lo32
  g1[6] = 0u;                                        // stride hi / dim1_stride
  g1[7] = 0u;
  asm volatile("tensor_load_to_lds %0, %1" :: "s"(g0), "s"(g1) : "memory");
}

// ---------------------------------------------------------------------------
// Weight conversion to f16 (Wq|Wk|Wv|Wo concatenated row-blocks, + Wout)
// ---------------------------------------------------------------------------
__global__ void k_cvt(const float* __restrict__ Wq, const float* __restrict__ Wk,
                      const float* __restrict__ Wv, const float* __restrict__ Wo,
                      const float* __restrict__ Wout,
                      half_t* __restrict__ wall, half_t* __restrict__ wouth) {
  int i = blockIdx.x * 256 + threadIdx.x;
  const int nw = DH * DM;
  if (i < nw) {
    wall[i]        = (half_t)Wq[i];
    wall[nw + i]   = (half_t)Wk[i];
    wall[2*nw + i] = (half_t)Wv[i];
    wall[3*nw + i] = (half_t)Wo[i];
    wouth[i]       = (half_t)Wout[i];
  }
}

// ---------------------------------------------------------------------------
// Fused RMSNorm (fp32 math, f16 output) + zi/zf dot products
// ---------------------------------------------------------------------------
__global__ void k_norm(const float* __restrict__ x, const float* __restrict__ nw,
                       const float* __restrict__ wi, const float* __restrict__ wf,
                       const float* __restrict__ wib, const float* __restrict__ wfb,
                       half_t* __restrict__ nh, float* __restrict__ zi,
                       float* __restrict__ zf) {
  __shared__ float red[8], redi[8], redf[8];
  __shared__ float bc;
  int row = blockIdx.x;
  int tid = threadIdx.x;
  int wid = tid >> 5, lid = tid & 31;
  const float* xr = x + (size_t)row * DM;
  float4 xv = reinterpret_cast<const float4*>(xr)[tid];
  float ss = xv.x*xv.x + xv.y*xv.y + xv.z*xv.z + xv.w*xv.w;
  for (int off = 16; off; off >>= 1) ss += __shfl_xor(ss, off, 32);
  if (lid == 0) red[wid] = ss;
  __syncthreads();
  if (tid == 0) {
    float s = 0.f;
    for (int i = 0; i < 8; i++) s += red[i];
    bc = rsqrtf(s / (float)DM + EPSV);
  }
  __syncthreads();
  float rr = bc;
  float4 nwv = reinterpret_cast<const float4*>(nw)[tid];
  float4 wiv = reinterpret_cast<const float4*>(wi)[tid];
  float4 wfv = reinterpret_cast<const float4*>(wf)[tid];
  float n0 = xv.x*rr*nwv.x, n1 = xv.y*rr*nwv.y;
  float n2 = xv.z*rr*nwv.z, n3 = xv.w*rr*nwv.w;
  half_t* nr = nh + (size_t)row * DM;
  int c = tid * 4;
  nr[c+0] = (half_t)n0; nr[c+1] = (half_t)n1;
  nr[c+2] = (half_t)n2; nr[c+3] = (half_t)n3;
  float pi = n0*wiv.x + n1*wiv.y + n2*wiv.z + n3*wiv.w;
  float pf = n0*wfv.x + n1*wfv.y + n2*wfv.z + n3*wfv.w;
  for (int off = 16; off; off >>= 1) {
    pi += __shfl_xor(pi, off, 32);
    pf += __shfl_xor(pf, off, 32);
  }
  if (lid == 0) { redi[wid] = pi; redf[wid] = pf; }
  __syncthreads();
  if (tid == 0) {
    float si = 0.f, sf = 0.f;
    for (int i = 0; i < 8; i++) { si += redi[i]; sf += redf[i]; }
    zi[row] = si + wib[0];
    zf[row] = sf + wfb[0];
  }
}

// ---------------------------------------------------------------------------
// Stabilizer scan -> log-gates li_t = zi - m_t, lf_t = zf + m_{t-1} - m_t.
// lf forced to 0 at t=0 (true f_p0=0 only multiplies zero initial state).
// ---------------------------------------------------------------------------
__global__ void k_mscan(const float* __restrict__ zi, const float* __restrict__ zf,
                        float* __restrict__ li, float* __restrict__ lf) {
  int b = threadIdx.x;
  if (b >= B_) return;
  float m = -INFINITY;
  for (int t = 0; t < T_; t++) {
    float a = zi[b*T_ + t], f = zf[b*T_ + t];
    float mt = fmaxf(f + m, a);
    li[b*T_ + t] = a - mt;
    lf[b*T_ + t] = (t == 0) ? 0.f : (f + m - mt);
    m = mt;
  }
}

// ---------------------------------------------------------------------------
// WMMA fragment loaders (CDNA5 ISA 7.12.2, f16 16x16x32)
// ---------------------------------------------------------------------------
__device__ inline v16h load_a_frag(const half_t* __restrict__ A, int lda,
                                   int m0, int k0, int lane) {
  int m  = m0 + (lane & 15);
  int kb = k0 + ((lane >> 4) << 3);
  const v8h* p0 = reinterpret_cast<const v8h*>(A + (size_t)m * lda + kb);
  const v8h* p1 = reinterpret_cast<const v8h*>(A + (size_t)m * lda + kb + 16);
  v8h lo = *p0, hi = *p1;
  v16h a;
#pragma unroll
  for (int i = 0; i < 8; i++) { a[i] = lo[i]; a[i+8] = hi[i]; }
  return a;
}

__device__ inline v16h load_b_frag(const half_t* __restrict__ W, int ldw,
                                   int n0, int k0, int lane) {
  int n  = n0 + (lane & 15);
  int kb = k0 + ((lane >> 4) << 3);
  const v8h* p = reinterpret_cast<const v8h*>(W + (size_t)n * ldw + kb);
  v8h lo = p[0], hi = p[2];
  v16h b;
#pragma unroll
  for (int i = 0; i < 8; i++) { b[i] = lo[i]; b[i+8] = hi[i]; }
  return b;
}

// ---------------------------------------------------------------------------
// Fused QKVO projection GEMM. q/k written as scaled f16, v/o as f32.
// ---------------------------------------------------------------------------
__global__ void __launch_bounds__(256) k_qkvo(
    const half_t* __restrict__ Ah, const half_t* __restrict__ Wall,
    const float* __restrict__ bo,
    half_t* __restrict__ qh, half_t* __restrict__ kh,
    float* __restrict__ vbuf, float* __restrict__ obuf) {
  int wave = blockIdx.x * (blockDim.x >> 5) + (threadIdx.x >> 5);
  int lane = threadIdx.x & 31;
  const int NGRP = (4 * DH) / 64;        // 32
  int mtile = wave / NGRP;
  int ngrp  = wave % NGRP;
  int m0 = mtile * 16, n0 = ngrp * 64;
  v8f acc[4] = {};
  for (int k0 = 0; k0 < DM; k0 += 32) {
    v16h a  = load_a_frag(Ah, DM, m0, k0, lane);
    v16h b0 = load_b_frag(Wall, DM, n0,      k0, lane);
    v16h b1 = load_b_frag(Wall, DM, n0 + 16, k0, lane);
    v16h b2 = load_b_frag(Wall, DM, n0 + 32, k0, lane);
    v16h b3 = load_b_frag(Wall, DM, n0 + 48, k0, lane);
    acc[0] = WMMA_F16(a, b0, acc[0]);
    acc[1] = WMMA_F16(a, b1, acc[1]);
    acc[2] = WMMA_F16(a, b2, acc[2]);
    acc[3] = WMMA_F16(a, b3, acc[3]);
  }
  int ncl   = lane & 15;
  int mbase = m0 + ((lane >> 4) << 3);
#pragma unroll
  for (int tt = 0; tt < 4; tt++) {
    int n = n0 + tt * 16 + ncl;
    int which = n >> 9;
    int col = n & (DH - 1);
#pragma unroll
    for (int r = 0; r < 8; r++) {
      size_t idx = (size_t)(mbase + r) * DH + col;
      float y = acc[tt][r];
      if (which == 0)      qh[idx]  = (half_t)(y * QK_SCALE);
      else if (which == 1) kh[idx]  = (half_t)(y * QK_SCALE);
      else if (which == 2) vbuf[idx] = y;
      else                 obuf[idx] = 1.f / (1.f + __expf(-(y + bo[col])));
    }
  }
}

// ---------------------------------------------------------------------------
// Chunked WMMA recurrence, double-buffered staging:
//   Q tile  : TDM tensor_load_to_lds (TENSORcnt), issued by wave 0
//   K tile  : per-lane async b128 copies (ASYNCcnt)
// Next chunk's transfers are issued before waiting on the current chunk
// (in-order completion => partial-count waits), hiding staging latency
// behind the previous chunk's WMMA work.
// ---------------------------------------------------------------------------
__global__ void __launch_bounds__(256, 1) k_recur(
    const half_t* __restrict__ qh, const half_t* __restrict__ kh,
    const float* __restrict__ vbuf, const float* __restrict__ obuf,
    const float* __restrict__ li,   const float* __restrict__ lf,
    half_t* __restrict__ Hh) {
  extern __shared__ char smem[];
  // Double buffers selected by integer offset (no pointer arrays: LDS
  // addrspacecast in a static initializer is unlinkable).
  half_t* QsBase = reinterpret_cast<half_t*>(smem);          // 2 x 32x512 f16
  half_t* KsBase = reinterpret_cast<half_t*>(smem + 65536);  // 2 x 32x512 f16
  half_t* Ch   = reinterpret_cast<half_t*>(smem + 131072);   // 32x512 f16
  half_t* Kth  = reinterpret_cast<half_t*>(smem + 163840);   // 512x32 f16
  half_t* Vt   = reinterpret_cast<half_t*>(smem + 196608);   // 32x32 f16
  half_t* Sh   = reinterpret_cast<half_t*>(smem + 198656);   // [j][s] f16
  float*  CqP  = reinterpret_cast<float*>(smem + 200704);    // 8x16x32
  float*  nbuf = reinterpret_cast<float*>(smem + 217088);    // 512 f32
  float*  colsum = reinterpret_cast<float*>(smem + 219136);  // 32
  float*  nqv    = colsum + 32;
  float*  denomL = colsum + 64;
  float*  bjL    = colsum + 96;
  float*  liL    = colsum + 128;                             // total 219776 B
  const int HBUF = LCH * DH;                                 // halves per buffer

  int b   = blockIdx.x >> 4;
  int rb  = blockIdx.x & 15;
  int r0  = rb * 32;
  int tid = threadIdx.x;
  int w   = tid >> 5;
  int lane = tid & 31;
  int wr = w & 1, wc = w >> 1;
  const size_t bb = (size_t)b * T_;

  v8f Ct[8] = {};
  nbuf[tid] = 0.f;
  nbuf[tid + 256] = 0.f;

  auto stage = [&](int c2) {
    size_t roff2 = (bb + (size_t)c2 * LCH) * (size_t)DH;
    half_t* Qd = QsBase + (c2 & 1) * HBUF;
    half_t* Kd = KsBase + (c2 & 1) * HBUF;
    if (w == 0)
      tdm_load_2d_f16((unsigned)(uintptr_t)Qd, qh + roff2, LCH, DH);
#pragma unroll
    for (int rnd = 0; rnd < 8; rnd++) {
      int oh = (rnd * 256 + tid) * 8;            // 16B per lane
      async_ld_b128(Kd + oh, kh + roff2 + oh);
    }
  };

  stage(0);                                      // prime the pipeline

  for (int ch = 0; ch < NCH; ch++) {
    int t0 = ch * LCH;
    const half_t* Qc = QsBase + (ch & 1) * HBUF;
    const half_t* Kc = KsBase + (ch & 1) * HBUF;
    const size_t rowoff = (bb + t0) * (size_t)DH;
    bool more = (ch + 1 < NCH);

    // ---- Phase A: issue next-chunk staging; scalars; V^T; C_prev copy ----
    if (more) stage(ch + 1);
    if (tid < 64) colsum[tid] = 0.f;             // zero colsum + nqv
    if (tid < 32) {
      float s = 0.f;
      for (int u = 0; u <= tid; u++) s += lf[bb + t0 + u];
      bjL[tid] = s;
      liL[tid] = li[bb + t0 + tid];
    }
    {
      int sidx = tid >> 3, rq = (tid & 7) * 4;
      float4 vv = *reinterpret_cast<const float4*>(
          vbuf + rowoff + (size_t)sidx * DH + r0 + rq);
      Vt[(rq+0)*LCH + sidx] = (half_t)vv.x;
      Vt[(rq+1)*LCH + sidx] = (half_t)vv.y;
      Vt[(rq+2)*LCH + sidx] = (half_t)vv.z;
      Vt[(rq+3)*LCH + sidx] = (half_t)vv.w;
    }
#pragma unroll
    for (int ct = 0; ct < 8; ct++) {
      int col = wc*128 + ct*16 + (lane & 15);
      int rbase = wr*16 + ((lane >> 4) << 3);
#pragma unroll
      for (int rr = 0; rr < 8; rr++)
        Ch[(size_t)(rbase + rr) * DH + col] = (half_t)Ct[ct][rr];
    }
    if (ch + 2 < NCH) {                          // warm L2 two chunks ahead
      __builtin_prefetch(kh + rowoff + (size_t)2 * LCH * DH + tid * 64, 0, 1);
      __builtin_prefetch(qh + rowoff + (size_t)2 * LCH * DH + tid * 64, 0, 1);
    }
    // wait current chunk's transfers (next chunk's may stay in flight)
    wait_async_le(more ? 8 : 0);
    if (w == 0) wait_tensor_le(more ? 1 : 0);
    __syncthreads();

    // ---- Phase B: waves 0-3: S = K^T Q GEMM; waves 4-7: K_hat + n.q ----
    if (w < 4) {
      int st = w & 1, jt = w >> 1;
      v8f acc = {};
      for (int k0 = 0; k0 < DH; k0 += 32) {
        v16h a  = load_a_frag(Kc, DH, st * 16, k0, lane);
        v16h bq = load_b_frag(Qc, DH, jt * 16, k0, lane);
        acc = WMMA_F16(a, bq, acc);
      }
      int sbase = st*16 + ((lane >> 4) << 3);
      int jl = jt*16 + (lane & 15);
#pragma unroll
      for (int rr = 0; rr < 8; rr++) {
        int sl = sbase + rr;
        float sv = 0.f;
        if (sl <= jl) {
          float wgt = __expf(liL[sl] + bjL[jl] - bjL[sl]);
          sv = acc[rr] * wgt;
          atomicAdd(&colsum[jl], sv);
        }
        Sh[jl * LCH + sl] = (half_t)sv;
      }
    } else {
      int t2 = tid - 128;
      float bL = bjL[LCH - 1];
      for (int s = 0; s < LCH; s++) {            // scaled transposed keys
        float wk = __expf(liL[s] + bL - bjL[s]);
        const half_t* kr = Kc + (size_t)s * DH + t2 * 4;
#pragma unroll
        for (int cc = 0; cc < 4; cc++)
          Kth[(t2*4 + cc) * LCH + s] = (half_t)((float)kr[cc] * wk);
      }
      int j = t2 >> 2, q4 = t2 & 3;              // n_prev . q_j
      const half_t* qr = Qc + (size_t)j * DH + q4 * 128;
      float sacc = 0.f;
      for (int blk = 0; blk < 16; blk++) {
        v8h qv = reinterpret_cast<const v8h*>(qr)[blk];
#pragma unroll
        for (int e = 0; e < 8; e++)
          sacc += (float)qv[e] * nbuf[q4*128 + blk*8 + e];
      }
      atomicAdd(&nqv[j], sacc);
    }
    __syncthreads();

    // ---- Phase C: readout GEMMs + state update ----
    {
      v8f accI[2] = {};
      for (int kk = 0; kk < 4; kk++) {
        int k0 = wc * 128 + kk * 32;
        v16h a  = load_a_frag(Ch, DH, wr * 16, k0, lane);
        v16h b0 = load_b_frag(Qc, DH, 0,  k0, lane);
        v16h b1 = load_b_frag(Qc, DH, 16, k0, lane);
        accI[0] = WMMA_F16(a, b0, accI[0]);
        accI[1] = WMMA_F16(a, b1, accI[1]);
      }
      float e0 = __expf(bjL[lane & 15]);
      float e1 = __expf(bjL[16 + (lane & 15)]);
#pragma unroll
      for (int rr = 0; rr < 8; rr++) { accI[0][rr] *= e0; accI[1][rr] *= e1; }
      v16h av = load_a_frag(Vt, LCH, wr * 16, 0, lane);
      if (wc == 0) {                             // intra-chunk (wave-uniform)
        v16h s0 = load_b_frag(Sh, LCH, 0,  0, lane);
        v16h s1 = load_b_frag(Sh, LCH, 16, 0, lane);
        accI[0] = WMMA_F16(av, s0, accI[0]);
        accI[1] = WMMA_F16(av, s1, accI[1]);
      }
#pragma unroll
      for (int jt = 0; jt < 2; jt++)
#pragma unroll
        for (int rr = 0; rr < 8; rr++)
          CqP[w*512 + (((lane>>4)<<3) + rr)*32 + jt*16 + (lane & 15)] =
              accI[jt][rr];
      // state update: C = F*C + V K_hat^T  (K-dim = 32, one WMMA per tile)
      float F = __expf(bjL[LCH - 1]);
#pragma unroll
      for (int ct = 0; ct < 8; ct++) {
        v16h bk = load_b_frag(Kth, LCH, wc*128 + ct*16, 0, lane);
#pragma unroll
        for (int rr = 0; rr < 8; rr++) Ct[ct][rr] *= F;
        Ct[ct] = WMMA_F16(av, bk, Ct[ct]);
      }
      // n update: n = F*n + rowsum(K_hat)
#pragma unroll
      for (int hh = 0; hh < 2; hh++) {
        int c = tid + hh * 256;
        float s = 0.f;
        for (int ss = 0; ss < LCH; ss++) s += (float)Kth[c*LCH + ss];
        nbuf[c] = F * nbuf[c] + s;
      }
      if (tid < 32)
        denomL[tid] = fmaxf(fabsf(__expf(bjL[tid]) * nqv[tid] + colsum[tid]),
                            1.0f);
    }
    __syncthreads();

    // ---- Phase D: reduce partials, apply gate, write h ----
    {
      int j = tid >> 3;
      int rq = (tid & 7) * 4;
      float dj = denomL[j];
      size_t ro = (bb + t0 + j) * (size_t)DH + r0 + rq;
      float4 o4 = *reinterpret_cast<const float4*>(obuf + ro);
      float oo[4] = {o4.x, o4.y, o4.z, o4.w};
#pragma unroll
      for (int rr = 0; rr < 4; rr++) {
        int r = rq + rr;
        int wrx = r >> 4, ml = r & 15;
        float cq = 0.f;
#pragma unroll
        for (int wcx = 0; wcx < 4; wcx++)
          cq += CqP[(wcx*2 + wrx)*512 + ml*32 + j];
        Hh[ro + rr] = (half_t)(oo[rr] * cq / dj);
      }
    }
  }
}

// ---------------------------------------------------------------------------
// Output GEMM with residual: out = x + h_f16 @ Wout^T
// ---------------------------------------------------------------------------
__global__ void __launch_bounds__(256) k_out(
    const half_t* __restrict__ Hh, const half_t* __restrict__ Wouth,
    const float* __restrict__ x, float* __restrict__ out) {
  int wave = blockIdx.x * (blockDim.x >> 5) + (threadIdx.x >> 5);
  int lane = threadIdx.x & 31;
  const int NGRP = DM / 64;              // 16
  int mtile = wave / NGRP;
  int ngrp  = wave % NGRP;
  int m0 = mtile * 16, n0 = ngrp * 64;
  v8f acc[4] = {};
  for (int k0 = 0; k0 < DH; k0 += 32) {
    v16h a  = load_a_frag(Hh, DH, m0, k0, lane);
    v16h b0 = load_b_frag(Wouth, DH, n0,      k0, lane);
    v16h b1 = load_b_frag(Wouth, DH, n0 + 16, k0, lane);
    v16h b2 = load_b_frag(Wouth, DH, n0 + 32, k0, lane);
    v16h b3 = load_b_frag(Wouth, DH, n0 + 48, k0, lane);
    acc[0] = WMMA_F16(a, b0, acc[0]);
    acc[1] = WMMA_F16(a, b1, acc[1]);
    acc[2] = WMMA_F16(a, b2, acc[2]);
    acc[3] = WMMA_F16(a, b3, acc[3]);
  }
  int ncl   = lane & 15;
  int mbase = m0 + ((lane >> 4) << 3);
#pragma unroll
  for (int tt = 0; tt < 4; tt++) {
    int n = n0 + tt * 16 + ncl;
#pragma unroll
    for (int r = 0; r < 8; r++) {
      size_t idx = (size_t)(mbase + r) * DM + n;
      out[idx] = x[idx] + acc[tt][r];
    }
  }
}

// ---------------------------------------------------------------------------
extern "C" void kernel_launch(void* const* d_in, const int* in_sizes, int n_in,
                              void* d_out, int out_size, void* d_ws, size_t ws_size,
                              hipStream_t stream) {
  (void)in_sizes; (void)n_in; (void)out_size; (void)ws_size;
  const float* x    = (const float*)d_in[0];
  const float* nw   = (const float*)d_in[1];
  const float* Wq   = (const float*)d_in[2];
  const float* Wk   = (const float*)d_in[3];
  const float* Wv   = (const float*)d_in[4];
  const float* wi_w = (const float*)d_in[5];
  const float* wi_b = (const float*)d_in[6];
  const float* wf_w = (const float*)d_in[7];
  const float* wf_b = (const float*)d_in[8];
  const float* Wo   = (const float*)d_in[9];
  const float* bo   = (const float*)d_in[10];
  const float* Wout = (const float*)d_in[11];
  float* out = (float*)d_out;

  char* ws = (char*)d_ws;
  size_t off = 0;
  auto alloc = [&](size_t bytes) -> void* {
    void* p = ws + off;
    off += (bytes + 255) & ~(size_t)255;
    return p;
  };
  half_t* wall  = (half_t*)alloc((size_t)4 * DH * DM * 2);
  half_t* wouth = (half_t*)alloc((size_t)DM * DH * 2);
  half_t* nh    = (half_t*)alloc((size_t)BT * DM * 2);
  half_t* qh    = (half_t*)alloc((size_t)BT * DH * 2);
  half_t* kh    = (half_t*)alloc((size_t)BT * DH * 2);
  float*  vbuf  = (float*)alloc((size_t)BT * DH * 4);
  float*  obuf  = (float*)alloc((size_t)BT * DH * 4);
  float*  zi    = (float*)alloc((size_t)BT * 4);
  float*  zf    = (float*)alloc((size_t)BT * 4);
  float*  lib   = (float*)alloc((size_t)BT * 4);
  float*  lfb   = (float*)alloc((size_t)BT * 4);
  half_t* Hh    = (half_t*)alloc((size_t)BT * DH * 2);

  const int RECUR_SMEM = 219776;
  (void)hipFuncSetAttribute((const void*)k_recur,
                            hipFuncAttributeMaxDynamicSharedMemorySize,
                            RECUR_SMEM);

  k_cvt<<<(DH * DM + 255) / 256, 256, 0, stream>>>(Wq, Wk, Wv, Wo, Wout,
                                                   wall, wouth);
  k_norm<<<BT, 256, 0, stream>>>(x, nw, wi_w, wf_w, wi_b, wf_b, nh, zi, zf);
  k_mscan<<<1, 32, 0, stream>>>(zi, zf, lib, lfb);
  k_qkvo<<<4096, 256, 0, stream>>>(nh, wall, bo, qh, kh, vbuf, obuf);
  k_recur<<<B_ * (DH / 32), 256, RECUR_SMEM, stream>>>(qh, kh, vbuf, obuf,
                                                       lib, lfb, Hh);
  k_out<<<2048, 256, 0, stream>>>(Hh, wouth, x, out);
}
